// AttentionBasedRouter_74869869904291
// MI455X (gfx1250) — matrix operand order repaired and checked by reading, not compiled, not run on hardware
//
#include <hip/hip_runtime.h>

typedef __attribute__((ext_vector_type(16))) __bf16 v16bf;
typedef __attribute__((ext_vector_type(8)))  __bf16 v8bf;
typedef __attribute__((ext_vector_type(4)))  __bf16 v4bf;
typedef __attribute__((ext_vector_type(8)))  float  v8f;
typedef __attribute__((ext_vector_type(4)))  float  v4f;
typedef __attribute__((ext_vector_type(4)))  unsigned int v4u;
typedef __attribute__((ext_vector_type(8)))  int    v8i;
typedef __attribute__((ext_vector_type(4)))  int    v4i;

#define DD   1024
#define EE   8
#define HH   8
#define HDD  128
#define MT   64                 // tokens per workgroup
#define NMT  4                  // 16-row tiles per wave (MT/16)
#define BPAN (1024 * 80)        // one B panel in LDS (1024 rows x 80B skewed)

static constexpr float LN_EPS = 1e-5f;
static constexpr float SCL    = 0.08838834764831845f; // 1/sqrt(128)

#if defined(__AMDGCN__) && __has_builtin(__builtin_amdgcn_tensor_load_to_lds)
#define HAVE_TDM 1
#else
#define HAVE_TDM 0
#endif

__device__ __forceinline__ v8f wmma_bf16(v16bf a, v16bf b, v8f c) {
  return __builtin_amdgcn_wmma_f32_16x16x32_bf16(false, a, false, b, (short)0, c, false, false);
}
__device__ __forceinline__ v16bf cat16(v8bf lo, v8bf hi) {
  return __builtin_shufflevector(lo, hi, 0,1,2,3,4,5,6,7,8,9,10,11,12,13,14,15);
}
// bf16 A/ctx tile: pad 4 DWORDs every 256 DWORDs -> row stride 2080 B
__device__ __forceinline__ int offBF(int m, int k) { return m * 2080 + (k << 1) + ((k >> 9) << 4); }

// --- TDM: B weight panel [1024 rows x 32 bf16] -> LDS rows skewed to 80 B ---
__device__ __forceinline__ void tdm_load_bpanel(const __bf16* Wb, unsigned lds_addr, int kc) {
#if HAVE_TDM
  unsigned long long ga = (unsigned long long)(uintptr_t)(Wb + kc * 32);
  v4u g0;
  g0[0] = 1u;                                                   // count=1
  g0[1] = lds_addr;
  g0[2] = (unsigned)ga;
  g0[3] = ((unsigned)(ga >> 32) & 0x01FFFFFFu) | 0x80000000u;   // addr[56:32] | type=2
  v8i g1;
  // data_size=2B | pad_enable | pad_interval=16dw (code 3) | pad_amount=4dw (code 3)
  g1[0] = (int)((1u << 16) | (1u << 20) | (3u << 22) | (3u << 25));
  g1[1] = (int)(1024u << 16);   // tensor_dim0 = 1024
  g1[2] = (int)(1024u << 16);   // tensor_dim1 = 1024
  g1[3] = (int)(32u << 16);     // tile_dim0 = 32
  g1[4] = 1024;                 // tile_dim1 = 1024, tile_dim2 = 0
  g1[5] = 1024;                 // tensor_dim0_stride = 1024
  g1[6] = 0; g1[7] = 0;
  v4i gz = {0, 0, 0, 0};
#if __has_include(<hip/amd_detail/amd_gfx1250_TDM.h>)
  v8i gz8 = {0, 0, 0, 0, 0, 0, 0, 0};
  __builtin_amdgcn_tensor_load_to_lds(g0, g1, gz, gz, gz8, 0);
#else
  __builtin_amdgcn_tensor_load_to_lds(g0, g1, gz, gz, 0);
#endif
#endif
}

__device__ __forceinline__ void load_bpanel_fallback(const __bf16* Wb, char* dst, int kc, int tid) {
  for (int n = tid; n < 1024; n += 256) {
    const v8bf* src = (const v8bf*)(Wb + (size_t)n * DD + kc * 32);
    *(v8bf*)(dst + n * 80)      = src[0];
    *(v8bf*)(dst + n * 80 + 16) = src[1];
  }
}

// One K=1024 GEMM: acc[mt][nt] += A(LDS bf16 tile) @ Wb^T, B panels via TDM.
__device__ __forceinline__ void gemm64(const __bf16* __restrict__ Wb,
                                       const char* at, char* bB,
                                       v8f acc[NMT][8],
                                       int tid, int col, int half, int nbase) {
#if HAVE_TDM
  if (tid < 32) {
    tdm_load_bpanel(Wb, (unsigned)(uintptr_t)(void*)bB, 0);
    __builtin_amdgcn_s_wait_tensorcnt(0);
  }
#else
  load_bpanel_fallback(Wb, bB, 0, tid);
#endif
  __syncthreads();

  for (int kc = 0; kc < 32; ++kc) {
    char* cur = bB + (size_t)(kc & 1) * BPAN;
    char* nxt = bB + (size_t)((kc + 1) & 1) * BPAN;
#if HAVE_TDM
    if (tid < 32 && kc + 1 < 32)
      tdm_load_bpanel(Wb, (unsigned)(uintptr_t)(void*)nxt, kc + 1);  // async
#endif
    v16bf af[NMT];
    #pragma unroll
    for (int mt = 0; mt < NMT; ++mt) {
      const v8bf* p0 = (const v8bf*)(at + offBF(mt * 16 + col, kc * 32 + half * 8));
      const v8bf* p1 = (const v8bf*)(at + offBF(mt * 16 + col, kc * 32 + 16 + half * 8));
      af[mt] = cat16(*p0, *p1);
    }
    #pragma unroll
    for (int nt = 0; nt < 8; ++nt) {
      const char* rp = cur + (nbase + nt * 16 + col) * 80 + 16 * half;
      v16bf bfr = cat16(*(const v8bf*)rp, *(const v8bf*)(rp + 32));
      #pragma unroll
      for (int mt = 0; mt < NMT; ++mt)
        acc[mt][nt] = wmma_bf16(af[mt], bfr, acc[mt][nt]);
    }
#if HAVE_TDM
    if (tid < 32 && kc + 1 < 32) __builtin_amdgcn_s_wait_tensorcnt(0);
    __syncthreads();
#else
    __syncthreads();
    if (kc + 1 < 32) { load_bpanel_fallback(Wb, nxt, kc + 1, tid); __syncthreads(); }
#endif
  }
}

// ---------------- weight fp32 -> bf16 (row-major [n][k]) ----------------
__global__ void wcvt(const float* __restrict__ W, __bf16* __restrict__ Wb, int n4) {
  int i = blockIdx.x * blockDim.x + threadIdx.x;
  if (i < n4) {
    v4f w = *(const v4f*)(W + (size_t)i * 4);
    *(v4bf*)(Wb + (size_t)i * 4) = __builtin_convertvector(w, v4bf);
  }
}

// ---------------- k/v expert projections (tiny: 2 x [8,1024]) ----------------
__global__ void kvproj(const float* __restrict__ emb,
                       const float* __restrict__ Wk, const float* __restrict__ Wv,
                       const float* __restrict__ bk, const float* __restrict__ bv,
                       float* __restrict__ kout, float* __restrict__ vout) {
  int i = blockIdx.x * blockDim.x + threadIdx.x;   // 0 .. 2*E*D-1
  int which = i / (EE * DD);
  int j = i % (EE * DD);
  int e = j / DD, n = j % DD;
  const float* W = which ? Wv : Wk;
  const float* b = which ? bv : bk;
  const float* er = emb + (size_t)e * DD;
  const float* wr = W + (size_t)n * DD;
  float s = 0.f;
  #pragma unroll 4
  for (int k = 0; k < DD; ++k) s += er[k] * wr[k];
  (which ? vout : kout)[j] = s + b[n];
}

// ---------------- fully fused router layer ----------------
// 256 threads = 8 waves; wave w owns head w. 64 tokens per workgroup.
// GEMM1 (x@Wq^T) -> in-register attention -> ctx to LDS (reusing the A tile)
// -> GEMM2 (ctx@Wo^T) -> residual + LN1 + steering + LN2 -> out.
__global__ __launch_bounds__(256) void router_fused(
    const float* __restrict__ x,
    const __bf16* __restrict__ Wqb, const __bf16* __restrict__ Wob,
    const float* __restrict__ bq, const float* __restrict__ bo,
    const float* __restrict__ kall, const float* __restrict__ vall,
    const float* __restrict__ steering,
    const float* __restrict__ g1, const float* __restrict__ b1,
    const float* __restrict__ g2, const float* __restrict__ b2,
    float* __restrict__ out) {
  __shared__ __align__(16) char atile[MT * 2080];     // x (bf16) then ctx (bf16)
  __shared__ __align__(16) char bpan[2 * BPAN];       // TDM double-buffered B panel
  __shared__ float gsum[MT][EE];
  __shared__ float rs1[MT], rq1[MT], rs2[MT], rq2[MT];

  const int tid  = threadIdx.x;
  const int lane = tid & 31;
  const int col  = lane & 15;
  const int half = lane >> 4;
  const int m0   = blockIdx.x * MT;
  const int nbase = (tid >> 5) * HDD;

  // zero reducers + stage x tile (fp32 -> bf16) into LDS
  #pragma unroll
  for (int i = tid; i < MT * EE; i += 256) ((float*)gsum)[i] = 0.f;
  if (tid < MT) { rs1[tid] = 0.f; rq1[tid] = 0.f; rs2[tid] = 0.f; rq2[tid] = 0.f; }
  for (int i = tid * 4; i < MT * DD; i += 256 * 4) {
    int m = i >> 10, kk = i & (DD - 1);
    v4f w = *(const v4f*)(x + (size_t)(m0 + m) * DD + kk);
    *(v4bf*)(atile + offBF(m, kk)) = __builtin_convertvector(w, v4bf);
  }
  __syncthreads();

  v8f acc[NMT][8];
  #pragma unroll
  for (int mt = 0; mt < NMT; ++mt)
    #pragma unroll
    for (int nt = 0; nt < 8; ++nt)
      acc[mt][nt] = (v8f){0.f,0.f,0.f,0.f,0.f,0.f,0.f,0.f};

  // ---- GEMM1: q = x @ Wq^T ----
  gemm64(Wqb, atile, bpan, acc, tid, col, half, nbase);

  #pragma unroll
  for (int nt = 0; nt < 8; ++nt) {
    float bqv = bq[nbase + nt * 16 + col];
    #pragma unroll
    for (int mt = 0; mt < NMT; ++mt)
      #pragma unroll
      for (int r = 0; r < 8; ++r) acc[mt][nt][r] += bqv;
  }

  // ---- attention over E=8 experts; ctx overwrites the LDS tile ----
  #pragma unroll
  for (int mt = 0; mt < NMT; ++mt) {
    float sc[8][EE];
    #pragma unroll
    for (int r = 0; r < 8; ++r)
      #pragma unroll
      for (int e = 0; e < EE; ++e) sc[r][e] = 0.f;
    #pragma unroll
    for (int nt = 0; nt < 8; ++nt) {
      int n = nbase + nt * 16 + col;
      #pragma unroll
      for (int e = 0; e < EE; ++e) {
        float kv = kall[e * DD + n];
        #pragma unroll
        for (int r = 0; r < 8; ++r) sc[r][e] += acc[mt][nt][r] * kv;
      }
    }
    #pragma unroll
    for (int off = 1; off < 16; off <<= 1)
      #pragma unroll
      for (int r = 0; r < 8; ++r)
        #pragma unroll
        for (int e = 0; e < EE; ++e)
          sc[r][e] += __shfl_xor(sc[r][e], off, 32);

    float attn[8][EE];
    #pragma unroll
    for (int r = 0; r < 8; ++r) {
      float mx = -1e30f;
      #pragma unroll
      for (int e = 0; e < EE; ++e) mx = fmaxf(mx, sc[r][e] * SCL);
      float se = 0.f;
      #pragma unroll
      for (int e = 0; e < EE; ++e) { attn[r][e] = __expf(sc[r][e] * SCL - mx); se += attn[r][e]; }
      float inv = 1.f / se;
      #pragma unroll
      for (int e = 0; e < EE; ++e) attn[r][e] *= inv;
    }
    if (col == 0) {
      #pragma unroll
      for (int r = 0; r < 8; ++r)
        #pragma unroll
        for (int e = 0; e < EE; ++e)
          atomicAdd(&gsum[mt * 16 + half * 8 + r][e], attn[r][e] * (1.f / HH));
    }
    #pragma unroll
    for (int nt = 0; nt < 8; ++nt) {
      int n = nbase + nt * 16 + col;
      float vv[EE];
      #pragma unroll
      for (int e = 0; e < EE; ++e) vv[e] = vall[e * DD + n];
      #pragma unroll
      for (int r = 0; r < 8; ++r) {
        float s = 0.f;
        #pragma unroll
        for (int e = 0; e < EE; ++e) s += attn[r][e] * vv[e];
        *(__bf16*)(atile + offBF(mt * 16 + half * 8 + r, n)) = (__bf16)s;
      }
    }
  }
  __syncthreads();

  // gating: softmax(mean over heads), kept in LDS
  if (tid < MT) {
    float g[EE];
    #pragma unroll
    for (int e = 0; e < EE; ++e) g[e] = gsum[tid][e];
    float mx = -1e30f;
    #pragma unroll
    for (int e = 0; e < EE; ++e) mx = fmaxf(mx, g[e]);
    float se = 0.f;
    #pragma unroll
    for (int e = 0; e < EE; ++e) { g[e] = __expf(g[e] - mx); se += g[e]; }
    float inv = 1.f / se;
    #pragma unroll
    for (int e = 0; e < EE; ++e) gsum[tid][e] = g[e] * inv;
  }
  __syncthreads();

  // ---- GEMM2: attended = ctx @ Wo^T ----
  #pragma unroll
  for (int mt = 0; mt < NMT; ++mt)
    #pragma unroll
    for (int nt = 0; nt < 8; ++nt)
      acc[mt][nt] = (v8f){0.f,0.f,0.f,0.f,0.f,0.f,0.f,0.f};
  gemm64(Wob, atile, bpan, acc, tid, col, half, nbase);

  // y = x + attended (+bo)
  #pragma unroll
  for (int nt = 0; nt < 8; ++nt) {
    int n = nbase + nt * 16 + col;
    float bov = bo[n];
    #pragma unroll
    for (int mt = 0; mt < NMT; ++mt)
      #pragma unroll
      for (int r = 0; r < 8; ++r)
        acc[mt][nt][r] += bov + x[(size_t)(m0 + mt * 16 + half * 8 + r) * DD + n];
  }

  // reduction 1
  #pragma unroll
  for (int mt = 0; mt < NMT; ++mt)
    #pragma unroll
    for (int r = 0; r < 8; ++r) {
      float s = 0.f, q = 0.f;
      #pragma unroll
      for (int nt = 0; nt < 8; ++nt) { float v = acc[mt][nt][r]; s += v; q += v * v; }
      #pragma unroll
      for (int off = 1; off < 16; off <<= 1) {
        s += __shfl_xor(s, off, 32);
        q += __shfl_xor(q, off, 32);
      }
      if (col == 0) {
        atomicAdd(&rs1[mt * 16 + half * 8 + r], s);
        atomicAdd(&rq1[mt * 16 + half * 8 + r], q);
      }
    }
  __syncthreads();

  // LN1 + steering
  #pragma unroll
  for (int nt = 0; nt < 8; ++nt) {
    int n = nbase + nt * 16 + col;
    float g1n = g1[n], b1n = b1[n];
    float st[EE];
    #pragma unroll
    for (int e = 0; e < EE; ++e) st[e] = steering[e * DD + n];
    #pragma unroll
    for (int mt = 0; mt < NMT; ++mt)
      #pragma unroll
      for (int r = 0; r < 8; ++r) {
        int ml = mt * 16 + half * 8 + r;
        float mu  = rs1[ml] * (1.f / DD);
        float var = rq1[ml] * (1.f / DD) - mu * mu;
        float inv = rsqrtf(var + LN_EPS);
        float x1 = (acc[mt][nt][r] - mu) * inv * g1n + b1n;
        float sv = 0.f;
        #pragma unroll
        for (int e = 0; e < EE; ++e) sv += gsum[ml][e] * st[e];
        acc[mt][nt][r] = x1 + sv;
      }
  }

  // reduction 2
  #pragma unroll
  for (int mt = 0; mt < NMT; ++mt)
    #pragma unroll
    for (int r = 0; r < 8; ++r) {
      float s = 0.f, q = 0.f;
      #pragma unroll
      for (int nt = 0; nt < 8; ++nt) { float v = acc[mt][nt][r]; s += v; q += v * v; }
      #pragma unroll
      for (int off = 1; off < 16; off <<= 1) {
        s += __shfl_xor(s, off, 32);
        q += __shfl_xor(q, off, 32);
      }
      if (col == 0) {
        atomicAdd(&rs2[mt * 16 + half * 8 + r], s);
        atomicAdd(&rq2[mt * 16 + half * 8 + r], q);
      }
    }
  __syncthreads();

  // LN2 + store
  #pragma unroll
  for (int nt = 0; nt < 8; ++nt) {
    int n = nbase + nt * 16 + col;
    float g2n = g2[n], b2n = b2[n];
    #pragma unroll
    for (int mt = 0; mt < NMT; ++mt)
      #pragma unroll
      for (int r = 0; r < 8; ++r) {
        int ml = mt * 16 + half * 8 + r;
        float mu  = rs2[ml] * (1.f / DD);
        float var = rq2[ml] * (1.f / DD) - mu * mu;
        float inv = rsqrtf(var + LN_EPS);
        out[(size_t)(m0 + mt * 16 + half * 8 + r) * DD + n] =
            (acc[mt][nt][r] - mu) * inv * g2n + b2n;
      }
  }
}

extern "C" void kernel_launch(void* const* d_in, const int* in_sizes, int n_in,
                              void* d_out, int out_size, void* d_ws, size_t ws_size,
                              hipStream_t stream) {
  const float* x        = (const float*)d_in[0];
  const float* emb      = (const float*)d_in[1];
  const float* steering = (const float*)d_in[2];
  const float* Wq       = (const float*)d_in[3];
  const float* Wk       = (const float*)d_in[4];
  const float* Wv       = (const float*)d_in[5];
  const float* bq       = (const float*)d_in[6];
  const float* bk       = (const float*)d_in[7];
  const float* bv       = (const float*)d_in[8];
  const float* Wo       = (const float*)d_in[9];
  const float* bo       = (const float*)d_in[10];
  const float* g1       = (const float*)d_in[11];
  const float* b1       = (const float*)d_in[12];
  const float* g2       = (const float*)d_in[13];
  const float* b2       = (const float*)d_in[14];
  float* out = (float*)d_out;

  const int ntok = in_sizes[0] / DD;          // 32768
  char* ws = (char*)d_ws;
  __bf16* Wqb  = (__bf16*)(ws);                                   // 2 MB
  __bf16* Wob  = (__bf16*)(ws + (size_t)(2 << 20));               // 2 MB
  float*  kall = (float*)(ws + (size_t)(4 << 20));                // 32 KB
  float*  vall = (float*)(ws + (size_t)(4 << 20) + 32768);        // 32 KB

  wcvt<<<(DD * DD / 4 + 255) / 256, 256, 0, stream>>>(Wq, Wqb, DD * DD / 4);
  wcvt<<<(DD * DD / 4 + 255) / 256, 256, 0, stream>>>(Wo, Wob, DD * DD / 4);
  kvproj<<<(2 * EE * DD) / 256, 256, 0, stream>>>(emb, Wk, Wv, bk, bv, kall, vall);
  router_fused<<<ntok / MT, 256, 0, stream>>>(x, Wqb, Wob, bq, bo, kall, vall,
                                              steering, g1, b1, g2, b2, out);
}
